// SeaformerAttention_85469849190684
// MI455X (gfx1250) — compile-verified, matrix operand order checked by reading, not run on hardware
//
#include <hip/hip_runtime.h>

// ---------------------------------------------------------------------------
// Seaformer attention block for MI455X (gfx1250, wave32, WMMA).
// All 1x1 convs run as v_wmma_f32_16x16x32_bf16 GEMMs with 16-pixel N-tiles.
// The qkv GEMM stages its activation tile via GLOBAL_LOAD_ASYNC_TO_LDS_B128
// (ASYNCcnt path) from a pre-transposed bf16 copy of x. dwconv3x3+relu is
// fused into the pointwise GEMM via an LDS bf16 tile, and the final
// relu(v+xr+xc) -> proj -> h_sigmoid * qkv path is fully fused.
//
// Input order assumption (setup_inputs() insertion order, tuples flattened):
//  0: hidden_states [16,128,80,80]
//  1: q_w  2-5: q_bn(g,b,m,v)   6: k_w  7-10: k_bn   11: v_w 12-15: v_bn
// 16: dw_w 17-20: dw_bn        21: pw_w 22-25: pw_bn 26: row_w 27-30: row_bn
// 31: col_w 32-35: col_bn      36: proj_w 37-40: proj_bn
// 41: pe_rowq 42: pe_rowk 43: pe_colq 44: pe_colk
// ---------------------------------------------------------------------------

typedef __attribute__((ext_vector_type(16))) __bf16 v16bf;
typedef __attribute__((ext_vector_type(8)))  float  v8f;

union BfFrag { v16bf v; unsigned int u[8]; };

__device__ __forceinline__ unsigned short f2bf(float f) {
  unsigned int u = __float_as_uint(f);
  unsigned int r = (u + 0x7FFFu + ((u >> 16) & 1u)) >> 16;  // round-nearest-even
  return (unsigned short)r;
}

// ------------------------- prep kernels ------------------------------------

__global__ void fold_w_kernel(const float* __restrict__ w,
                              const float* __restrict__ gamma,
                              const float* __restrict__ var,
                              unsigned short* __restrict__ dst,
                              int rows, int cols) {
  int i = blockIdx.x * blockDim.x + threadIdx.x;
  if (i >= rows * cols) return;
  int o = i / cols;
  float inv = gamma[o] * rsqrtf(var[o] + 1e-5f);
  dst[i] = f2bf(w[i] * inv);
}

__global__ void fold_b_kernel(const float* __restrict__ gamma,
                              const float* __restrict__ beta,
                              const float* __restrict__ mean,
                              const float* __restrict__ var,
                              float* __restrict__ dst, int n) {
  int i = blockIdx.x * blockDim.x + threadIdx.x;
  if (i >= n) return;
  float inv = gamma[i] * rsqrtf(var[i] + 1e-5f);
  dst[i] = beta[i] - mean[i] * inv;
}

__global__ void fold_dw_kernel(const float* __restrict__ w,   // [512*9]
                               const float* __restrict__ gamma,
                               const float* __restrict__ var,
                               float* __restrict__ dst) {
  int i = blockIdx.x * blockDim.x + threadIdx.x;
  if (i >= 512 * 9) return;
  int c = i / 9;
  dst[i] = w[i] * gamma[c] * rsqrtf(var[c] + 1e-5f);
}

__global__ void pe_interp_kernel(const float* __restrict__ pe,   // [128][16]
                                 float* __restrict__ dst) {      // [128][80]
  int i = blockIdx.x * blockDim.x + threadIdx.x;
  if (i >= 128 * 80) return;
  int c = i / 80, x = i % 80;
  float xs = ((float)x + 0.5f) * (16.0f / 80.0f) - 0.5f;
  float x0 = floorf(xs);
  float wg = xs - x0;
  int i0 = min(max((int)x0, 0), 15);
  int i1 = min(max((int)x0 + 1, 0), 15);
  dst[i] = pe[c * 16 + i0] * (1.0f - wg) + pe[c * 16 + i1] * wg;
}

// Transpose x [16][128][6400] f32 -> xt bf16, tiled [b*400+tile][pix16][c128]
// so each 16-pixel x 128-ch GEMM tile is 4KB contiguous for async LDS DMA.
__global__ __launch_bounds__(256) void xpose_kernel(
    const float* __restrict__ x, unsigned short* __restrict__ xt) {
  const long i = (long)blockIdx.x * 256 + threadIdx.x;  // one thread = 8 elems
  const long base = i * 8;
  if (base >= (long)16 * 128 * 6400) return;
  const int s = (int)(base % 6400);
  const long bc = base / 6400;
  const int c = (int)(bc % 128);
  const int b = (int)(bc / 128);
  const float* src = x + base;
  const float4 v0 = *(const float4*)(src);
  const float4 v1 = *(const float4*)(src + 4);
  const float vv[8] = {v0.x, v0.y, v0.z, v0.w, v1.x, v1.y, v1.z, v1.w};
#pragma unroll
  for (int j = 0; j < 8; j++) {
    const int ss = s + j;
    const int tile = ss >> 4, pix = ss & 15;
    xt[(((long)b * 400 + tile) * 16 + pix) * 128 + c] = f2bf(vv[j]);
  }
}

// ------------------------- generic WMMA tile GEMM --------------------------
// out[b, o, s] = sum_c maybe_relu(in[b, c, s]) * Wb[o, c]  + bias[o]
// One WG = 16 pixels (contiguous s) x OC out-channels.  8 waves, each wave
// owns OC/8 out-channels as (OC/128) 16x16 WMMA accumulator tiles.
// ASYNC=true: `in` is the pre-transposed bf16 tile buffer; the activation
// tile is staged with GLOBAL_LOAD_ASYNC_TO_LDS_B128 (tracked by ASYNCcnt).

template <int K, int OC, bool RELU, bool ASYNC>
__global__ __launch_bounds__(256) void gemm_tile_kernel(
    const float* __restrict__ in, long in_bs, int L, int tiles_per_batch,
    const unsigned short* __restrict__ Wb, const float* __restrict__ bias,
    float* __restrict__ out, long out_bs) {
  constexpr int TPW = OC / 128;  // accumulator tiles per wave
  static_assert(!ASYNC || K == 128, "async staging sized for K=128");
  // +8 shorts (16B) row padding: keeps async B128 LDS writes 16B-aligned and
  // spreads the b32 fragment gathers across banks (row = 68 dwords).
  __shared__ unsigned short lds[16][K + 8];
  const int bid = blockIdx.x;
  const int b = bid / tiles_per_batch;
  const int s0 = (bid % tiles_per_batch) * 16;
  const int t = threadIdx.x;

  // Prime weight matrix into cache (global_prefetch_b8).
  __builtin_prefetch(Wb + (size_t)t * (OC * K / 256), 0, 0);

  if (ASYNC) {
    // 256 threads x 16B = the whole 16x128 bf16 tile, DMA'd straight to LDS.
    const unsigned short* tb = (const unsigned short*)in + (size_t)bid * 16 * K;
    const int pix = t >> 4, chunk = t & 15;
    const unsigned long long g =
        (unsigned long long)(tb + (size_t)pix * K + chunk * 8);
    const unsigned lo = (unsigned)(pix * (K + 8) * 2 + chunk * 16);
    asm volatile("global_load_async_to_lds_b128 %0, %1, off"
                 :: "v"(lo), "v"(g) : "memory");
    asm volatile("s_wait_asynccnt 0x0" ::: "memory");
  } else {
    const float* inb = in + (long)b * in_bs;
    for (int task = t; task < K * 4; task += 256) {
      const int c = task >> 2;
      const int g = task & 3;
      const float4 v4 = *(const float4*)(inb + (long)c * L + s0 + g * 4);
      float x0 = v4.x, x1 = v4.y, x2 = v4.z, x3 = v4.w;
      if (RELU) {
        x0 = fmaxf(x0, 0.f); x1 = fmaxf(x1, 0.f);
        x2 = fmaxf(x2, 0.f); x3 = fmaxf(x3, 0.f);
      }
      lds[g * 4 + 0][c] = f2bf(x0);
      lds[g * 4 + 1][c] = f2bf(x1);
      lds[g * 4 + 2][c] = f2bf(x2);
      lds[g * 4 + 3][c] = f2bf(x3);
    }
  }
  __syncthreads();

  const int wave = t >> 5, lane = t & 31;
  const int mrow = lane & 15;          // A-matrix row / B-matrix column
  const int kb = (lane >> 4) << 3;     // per-lane K sub-block

  v8f zero = {};
  v8f acc[TPW];
#pragma unroll
  for (int m = 0; m < TPW; m++) acc[m] = zero;

#pragma unroll
  for (int ks = 0; ks < K / 32; ks++) {
    BfFrag bf;
#pragma unroll
    for (int vg = 0; vg < 8; vg++) {
      const int k = ks * 32 + ((vg & 4) << 2) + kb + ((vg & 3) << 1);
      bf.u[vg] = *(const unsigned int*)&lds[mrow][k];
    }
#pragma unroll
    for (int m = 0; m < TPW; m++) {
      const int och = wave * (TPW * 16) + m * 16 + mrow;
      const unsigned short* wr = Wb + (long)och * K + ks * 32;
      BfFrag af;
#pragma unroll
      for (int vg = 0; vg < 8; vg++) {
        const int k = ((vg & 4) << 2) + kb + ((vg & 3) << 1);
        af.u[vg] = *(const unsigned int*)(wr + k);
      }
      acc[m] = __builtin_amdgcn_wmma_f32_16x16x32_bf16(
          false, af.v, false, bf.v, (short)0, acc[m], false, false);
    }
  }

  const int pix = lane & 15;
  const int rofs = (lane >> 4) << 3;
  float* outb = out + (long)b * out_bs;
#pragma unroll
  for (int m = 0; m < TPW; m++) {
    const int obase = wave * (TPW * 16) + m * 16;
#pragma unroll
    for (int r = 0; r < 8; r++) {
      const int och = obase + r + rofs;
      outb[(long)och * L + s0 + pix] = acc[m][r] + bias[och];
    }
  }
}

// ------------------------- row / col means ---------------------------------

__global__ __launch_bounds__(160) void means_kernel(
    const float* __restrict__ qkv, float* __restrict__ rowmean,
    float* __restrict__ colmean) {
  const int bc = blockIdx.x;  // b*512 + c
  const float* plane = qkv + (long)bc * 6400;
  const int t = threadIdx.x;
  if (t < 80) {
    float s = 0.f;
    const float* r = plane + t * 80;
    for (int w = 0; w < 80; w++) s += r[w];
    rowmean[(long)bc * 80 + t] = s * (1.0f / 80.0f);
  } else if (t < 160) {
    const int w = t - 80;
    float s = 0.f;
    for (int h = 0; h < 80; h++) s += plane[h * 80 + w];
    colmean[(long)bc * 80 + w] = s * (1.0f / 80.0f);
  }
}

// ------------------------- fused dwconv3x3 + pwconv ------------------------

__global__ __launch_bounds__(256) void dwpw_kernel(
    const float* __restrict__ qkv, const float* __restrict__ dww,  // [512][9]
    const float* __restrict__ dwb, const unsigned short* __restrict__ Wpw,
    const float* __restrict__ pwb, float* __restrict__ pwout) {
  __shared__ unsigned short lds[16][516];  // [pixel][512ch], padded
  const int bid = blockIdx.x;
  const int b = bid / 400;
  const int s0 = (bid % 400) * 16;
  const int t = threadIdx.x;
  const float* qb = qkv + (long)b * 512 * 6400;

  __builtin_prefetch(Wpw + (size_t)t * 256, 0, 0);  // 128x512 bf16 weights

  // Phase 1: depthwise 3x3 + BN + relu for 512 ch x 16 pixels -> LDS bf16.
  for (int task = t; task < 512 * 16; task += 256) {
    const int c = task >> 4;
    const int pix = task & 15;
    const int s = s0 + pix;
    const int h = s / 80, w = s % 80;
    const float* pl = qb + (long)c * 6400;
    const float* wc = dww + c * 9;
    float acc = 0.f;
#pragma unroll
    for (int kh = 0; kh < 3; kh++) {
      const int hh = h + kh - 1;
      if (hh < 0 || hh >= 80) continue;
#pragma unroll
      for (int kw = 0; kw < 3; kw++) {
        const int ww = w + kw - 1;
        if (ww < 0 || ww >= 80) continue;
        acc += pl[hh * 80 + ww] * wc[kh * 3 + kw];
      }
    }
    acc = fmaxf(acc + dwb[c], 0.f);
    lds[pix][c] = f2bf(acc);
  }
  __syncthreads();

  // Phase 2: pw GEMM  OC=128, K=512  (each wave owns one 16x16 tile).
  const int wave = t >> 5, lane = t & 31;
  const int mrow = lane & 15;
  const int kb = (lane >> 4) << 3;
  v8f acc = {};
#pragma unroll
  for (int ks = 0; ks < 16; ks++) {
    BfFrag bf;
#pragma unroll
    for (int vg = 0; vg < 8; vg++) {
      const int k = ks * 32 + ((vg & 4) << 2) + kb + ((vg & 3) << 1);
      bf.u[vg] = *(const unsigned int*)&lds[mrow][k];
    }
    const int och = wave * 16 + mrow;
    const unsigned short* wr = Wpw + (long)och * 512 + ks * 32;
    BfFrag af;
#pragma unroll
    for (int vg = 0; vg < 8; vg++) {
      const int k = ((vg & 4) << 2) + kb + ((vg & 3) << 1);
      af.u[vg] = *(const unsigned int*)(wr + k);
    }
    acc = __builtin_amdgcn_wmma_f32_16x16x32_bf16(
        false, af.v, false, bf.v, (short)0, acc, false, false);
  }
  const int pix = lane & 15;
  const int rofs = (lane >> 4) << 3;
#pragma unroll
  for (int r = 0; r < 8; r++) {
    const int och = wave * 16 + r + rofs;
    pwout[((long)b * 128 + och) * 6400 + s0 + pix] = acc[r] + pwb[och];
  }
}

// ------------------------- axial attention ---------------------------------

__global__ __launch_bounds__(128) void axial_attn_kernel(
    const float* __restrict__ rowmean, const float* __restrict__ colmean,
    const float* __restrict__ pe4,  // [4][128][80]: rowq,rowk,colq,colk
    float* __restrict__ xr_att, float* __restrict__ xc_att) {
  const int bid = blockIdx.x;  // rc*128 + b*8 + head
  const int rc = bid >> 7;
  const int b = (bid & 127) >> 3;
  const int head = bid & 7;
  const float* mean = rc ? colmean : rowmean;
  const float* peq = pe4 + (rc ? 2 : 0) * 128 * 80;
  const float* pek = pe4 + (rc ? 3 : 1) * 128 * 80;
  float* outp = (rc ? xc_att : xr_att) + ((long)b * 256 + head * 32) * 80;

  __shared__ float qs[80][16];
  __shared__ float ksm[80][16];
  __shared__ float vs[80][32];
  const int t = threadIdx.x;
  const float* mb = mean + (long)b * 512 * 80;

  for (int e = t; e < 80 * 16; e += 128) {
    const int i = e & 15, l = e >> 4;
    const int cq = head * 16 + i;
    qs[l][i] = mb[cq * 80 + l] + peq[cq * 80 + l];
    ksm[l][i] = mb[(128 + cq) * 80 + l] + pek[cq * 80 + l];
  }
  for (int e = t; e < 80 * 32; e += 128) {
    const int d = e & 31, l = e >> 5;
    vs[l][d] = mb[(256 + head * 32 + d) * 80 + l];
  }
  __syncthreads();

  if (t < 80) {
    float q[16];
#pragma unroll
    for (int i = 0; i < 16; i++) q[i] = qs[t][i];
    float mx = -1e30f;
    for (int j = 0; j < 80; j++) {
      float dd = 0.f;
#pragma unroll
      for (int i = 0; i < 16; i++) dd += q[i] * ksm[j][i];
      mx = fmaxf(mx, dd * 0.25f);
    }
    float sum = 0.f;
    float o[32];
#pragma unroll
    for (int d = 0; d < 32; d++) o[d] = 0.f;
    for (int j = 0; j < 80; j++) {
      float dd = 0.f;
#pragma unroll
      for (int i = 0; i < 16; i++) dd += q[i] * ksm[j][i];
      const float e = __expf(dd * 0.25f - mx);
      sum += e;
#pragma unroll
      for (int d = 0; d < 32; d++) o[d] += e * vs[j][d];
    }
    const float inv = 1.0f / sum;
    for (int d = 0; d < 32; d++) outp[d * 80 + t] = o[d] * inv;
  }
}

// ------------------------- final fused kernel ------------------------------
// t = relu(v + xr + xc); proj GEMM (128x256); h_sigmoid(.) * pwout -> d_out

__global__ __launch_bounds__(256) void fuse_kernel(
    const float* __restrict__ qkv, const float* __restrict__ xr_fin,
    const float* __restrict__ xc_fin, const unsigned short* __restrict__ Wproj,
    const float* __restrict__ projb, const float* __restrict__ pwout,
    float* __restrict__ out) {
  __shared__ unsigned short lds[16][260];
  const int bid = blockIdx.x;
  const int b = bid / 400;
  const int s0 = (bid % 400) * 16;
  const int t = threadIdx.x;
  const float* vb = qkv + ((long)b * 512 + 256) * 6400;
  const float* xrb = xr_fin + (long)b * 256 * 80;
  const float* xcb = xc_fin + (long)b * 256 * 80;

  __builtin_prefetch(Wproj + (size_t)t * 128, 0, 0);  // 128x256 bf16 weights

  for (int task = t; task < 256 * 16; task += 256) {
    const int c = task >> 4;
    const int pix = task & 15;
    const int s = s0 + pix;
    const int h = s / 80, w = s % 80;
    const float val = vb[(long)c * 6400 + s] + xrb[c * 80 + h] + xcb[c * 80 + w];
    lds[pix][c] = f2bf(fmaxf(val, 0.f));
  }
  __syncthreads();

  const int wave = t >> 5, lane = t & 31;
  const int mrow = lane & 15;
  const int kb = (lane >> 4) << 3;
  v8f acc = {};
#pragma unroll
  for (int ks = 0; ks < 8; ks++) {
    BfFrag bf;
#pragma unroll
    for (int vg = 0; vg < 8; vg++) {
      const int k = ks * 32 + ((vg & 4) << 2) + kb + ((vg & 3) << 1);
      bf.u[vg] = *(const unsigned int*)&lds[mrow][k];
    }
    const int och = wave * 16 + mrow;
    const unsigned short* wr = Wproj + (long)och * 256 + ks * 32;
    BfFrag af;
#pragma unroll
    for (int vg = 0; vg < 8; vg++) {
      const int k = ((vg & 4) << 2) + kb + ((vg & 3) << 1);
      af.u[vg] = *(const unsigned int*)(wr + k);
    }
    acc = __builtin_amdgcn_wmma_f32_16x16x32_bf16(
        false, af.v, false, bf.v, (short)0, acc, false, false);
  }
  const int pix = lane & 15;
  const int rofs = (lane >> 4) << 3;
#pragma unroll
  for (int r = 0; r < 8; r++) {
    const int och = wave * 16 + r + rofs;
    const long idx = ((long)b * 128 + och) * 6400 + s0 + pix;
    const float xxv = acc[r] + projb[och];
    const float hs = fminf(fmaxf(xxv + 3.0f, 0.0f), 6.0f) * (1.0f / 6.0f);
    out[idx] = hs * pwout[idx];
  }
}

// ------------------------- host-side launcher ------------------------------

extern "C" void kernel_launch(void* const* d_in, const int* in_sizes, int n_in,
                              void* d_out, int out_size, void* d_ws,
                              size_t ws_size, hipStream_t stream) {
  const float* x = (const float*)d_in[0];
  const float* q_w = (const float*)d_in[1];
  const float *q_g = (const float*)d_in[2], *q_b = (const float*)d_in[3],
              *q_m = (const float*)d_in[4], *q_v = (const float*)d_in[5];
  const float* k_w = (const float*)d_in[6];
  const float *k_g = (const float*)d_in[7], *k_b = (const float*)d_in[8],
              *k_m = (const float*)d_in[9], *k_v = (const float*)d_in[10];
  const float* v_w = (const float*)d_in[11];
  const float *v_g = (const float*)d_in[12], *v_b = (const float*)d_in[13],
              *v_m = (const float*)d_in[14], *v_v = (const float*)d_in[15];
  const float* dw_w = (const float*)d_in[16];
  const float *dw_g = (const float*)d_in[17], *dw_b = (const float*)d_in[18],
              *dw_m = (const float*)d_in[19], *dw_v = (const float*)d_in[20];
  const float* pw_w = (const float*)d_in[21];
  const float *pw_g = (const float*)d_in[22], *pw_b = (const float*)d_in[23],
              *pw_m = (const float*)d_in[24], *pw_v = (const float*)d_in[25];
  const float* row_w = (const float*)d_in[26];
  const float *row_g = (const float*)d_in[27], *row_b = (const float*)d_in[28],
              *row_m = (const float*)d_in[29], *row_v = (const float*)d_in[30];
  const float* col_w = (const float*)d_in[31];
  const float *col_g = (const float*)d_in[32], *col_b = (const float*)d_in[33],
              *col_m = (const float*)d_in[34], *col_v = (const float*)d_in[35];
  const float* proj_w = (const float*)d_in[36];
  const float *proj_g = (const float*)d_in[37], *proj_b = (const float*)d_in[38],
              *proj_m = (const float*)d_in[39], *proj_v = (const float*)d_in[40];
  const float* pe_rowq = (const float*)d_in[41];
  const float* pe_rowk = (const float*)d_in[42];
  const float* pe_colq = (const float*)d_in[43];
  const float* pe_colk = (const float*)d_in[44];

  char* ws = (char*)d_ws;
  // -- small parameter area --
  unsigned short* Wqkv = (unsigned short*)(ws + 0);         // 512x128 bf16
  unsigned short* Wpw = (unsigned short*)(ws + 131072);     // 128x512
  unsigned short* Wrow = (unsigned short*)(ws + 262144);    // 256x256
  unsigned short* Wcol = (unsigned short*)(ws + 393216);    // 256x256
  unsigned short* Wproj = (unsigned short*)(ws + 524288);   // 128x256
  float* b_qkv = (float*)(ws + 589824);                     // 512
  float* b_pw = (float*)(ws + 591872);                      // 128
  float* b_row = (float*)(ws + 592384);                     // 256
  float* b_col = (float*)(ws + 593408);                     // 256
  float* b_proj = (float*)(ws + 594432);                    // 128
  float* dww = (float*)(ws + 594944);                       // 512x9
  float* dwb = (float*)(ws + 613376);                       // 512
  float* pe4 = (float*)(ws + 615424);                       // 4x128x80
  // -- big buffers --
  float* qkv_buf = (float*)(ws + 1048576u);                 // 16x512x6400 f32
  float* pw_out = (float*)(ws + 211812352u);                // 16x128x6400 f32
  // x_t (26MB, bf16 tiles) aliases pw_out: consumed by the qkv GEMM before
  // dwpw_kernel produces pw_out (stream-ordered, no overlap in lifetime).
  unsigned short* x_t = (unsigned short*)(ws + 211812352u);
  float* rowmean = (float*)(ws + 264241152u);               // 16x512x80
  float* colmean = (float*)(ws + 266862592u);               // 16x512x80
  float* xr_att = (float*)(ws + 269484032u);                // 16x256x80
  float* xc_att = (float*)(ws + 270794752u);                // 16x256x80
  float* xr_fin = (float*)(ws + 272105472u);                // 16x256x80
  float* xc_fin = (float*)(ws + 273416192u);                // 16x256x80

  // ---- prep: fold BN into weights/biases, interpolate PEs ----
  fold_w_kernel<<<64, 256, 0, stream>>>(q_w, q_g, q_v, Wqkv, 128, 128);
  fold_w_kernel<<<64, 256, 0, stream>>>(k_w, k_g, k_v, Wqkv + 16384, 128, 128);
  fold_w_kernel<<<128, 256, 0, stream>>>(v_w, v_g, v_v, Wqkv + 32768, 256, 128);
  fold_w_kernel<<<256, 256, 0, stream>>>(pw_w, pw_g, pw_v, Wpw, 128, 512);
  fold_w_kernel<<<256, 256, 0, stream>>>(row_w, row_g, row_v, Wrow, 256, 256);
  fold_w_kernel<<<256, 256, 0, stream>>>(col_w, col_g, col_v, Wcol, 256, 256);
  fold_w_kernel<<<128, 256, 0, stream>>>(proj_w, proj_g, proj_v, Wproj, 128, 256);
  fold_b_kernel<<<1, 256, 0, stream>>>(q_g, q_b, q_m, q_v, b_qkv, 128);
  fold_b_kernel<<<1, 256, 0, stream>>>(k_g, k_b, k_m, k_v, b_qkv + 128, 128);
  fold_b_kernel<<<1, 256, 0, stream>>>(v_g, v_b, v_m, v_v, b_qkv + 256, 256);
  fold_b_kernel<<<1, 256, 0, stream>>>(pw_g, pw_b, pw_m, pw_v, b_pw, 128);
  fold_b_kernel<<<1, 256, 0, stream>>>(row_g, row_b, row_m, row_v, b_row, 256);
  fold_b_kernel<<<1, 256, 0, stream>>>(col_g, col_b, col_m, col_v, b_col, 256);
  fold_b_kernel<<<1, 256, 0, stream>>>(proj_g, proj_b, proj_m, proj_v, b_proj, 128);
  fold_dw_kernel<<<18, 256, 0, stream>>>(dw_w, dw_g, dw_v, dww);
  fold_b_kernel<<<2, 256, 0, stream>>>(dw_g, dw_b, dw_m, dw_v, dwb, 512);
  pe_interp_kernel<<<40, 256, 0, stream>>>(pe_rowq, pe4 + 0 * 10240);
  pe_interp_kernel<<<40, 256, 0, stream>>>(pe_rowk, pe4 + 1 * 10240);
  pe_interp_kernel<<<40, 256, 0, stream>>>(pe_colq, pe4 + 2 * 10240);
  pe_interp_kernel<<<40, 256, 0, stream>>>(pe_colk, pe4 + 3 * 10240);

  // ---- transpose x to bf16 tiles for async LDS staging ----
  xpose_kernel<<<6400, 256, 0, stream>>>(x, x_t);

  // ---- q/k/v fused 1x1 conv GEMM: x_t -> qkv[16,512,6400] ----
  gemm_tile_kernel<128, 512, false, true><<<6400, 256, 0, stream>>>(
      (const float*)x_t, 0, 6400, 400, Wqkv, b_qkv, qkv_buf, (long)512 * 6400);

  // ---- row/col means of q,k,v ----
  means_kernel<<<16 * 512, 160, 0, stream>>>(qkv_buf, rowmean, colmean);

  // ---- fused dwconv3x3+BN+relu -> pwconv(512->128) ----
  dwpw_kernel<<<6400, 256, 0, stream>>>(qkv_buf, dww, dwb, Wpw, b_pw, pw_out);

  // ---- axial attention (row rc=0, col rc=1) ----
  axial_attn_kernel<<<256, 128, 0, stream>>>(rowmean, colmean, pe4, xr_att,
                                             xc_att);

  // ---- row_w / col_w 256x256 convs with relu-on-input ----
  gemm_tile_kernel<256, 256, true, false><<<80, 256, 0, stream>>>(
      xr_att, (long)256 * 80, 80, 5, Wrow, b_row, xr_fin, (long)256 * 80);
  gemm_tile_kernel<256, 256, true, false><<<80, 256, 0, stream>>>(
      xc_att, (long)256 * 80, 80, 5, Wcol, b_col, xc_fin, (long)256 * 80);

  // ---- final fuse: relu(v+xr+xc) -> proj -> h_sigmoid * pw_out ----
  fuse_kernel<<<6400, 256, 0, stream>>>(qkv_buf, xr_fin, xc_fin, Wproj, b_proj,
                                        pw_out, (float*)d_out);
}
